// MambaTCM_66133906424254
// MI455X (gfx1250) — matrix operand order repaired and checked by reading, not compiled, not run on hardware
//
#include <hip/hip_runtime.h>
#include <hip/hip_bf16.h>
#include <math.h>

// ---------------------------------------------------------------------------
// MambaTCM pipeline for gfx1250 (MI455X), wave32 + v_wmma_f32_16x16x32_bf16.
//
//   K1 backbone : implicit-GEMM Conv1d(2->256,k16,s8) + ReLU + mean  (WMMA)
//   K2 xz       : feat(2048x256) @ W_in(256x1024)       (WMMA, LDS-staged B)
//   K3 dconv    : causal depthwise conv(4) + bias + silu             (VALU)
//   K4 proj     : xm(2048x512) @ W_xp(512x48)           (WMMA, LDS-staged B)
//   K5 dt       : softplus(proj[:, :16] @ W_dt + b_dt)  (K=16, VALU)
//   K6 scan     : selective scan, 16-lane half-wave per (b,d), shfl  (VALU)
//   K7 gate     : (y + xm*D) * silu(z)                               (VALU)
//   K8 hidden   : y(2048x512) @ W_out(512x256)          (WMMA, LDS-staged B)
//   K9 cls      : hidden @ W_cls(256x5) + b_cls                      (VALU)
// ---------------------------------------------------------------------------

typedef __attribute__((ext_vector_type(16))) __bf16 v16bf;
typedef __attribute__((ext_vector_type(8)))  float  v8f;

#define D_MODEL   256
#define D_INNER   512
#define D_STATE   16
#define DT_RANK   16
#define N_CLS     5
#define C_IN      2
#define K_BB      16
#define STRIDE_BB 8
#define BATCH     4
#define SEQ       512
#define BS        2048          // BATCH*SEQ samples through the backbone
#define T_IN      3000
#define T_OUT     374           // (3000-16)/8 + 1
#define NPROJ     (DT_RANK + 2*D_STATE)   // 48

// ---------------- WMMA A-fragment loader (bf16, 16x16x32) ------------------
// A (16x32, MxK) row-major fp32 source, lda = row stride.
// lane<16:  M=lane, elems 0..15 -> K 0..15
// lane>=16: M=lane-16, elems 0..15 -> K 16..31
// Per lane this is one contiguous 16-float chunk -> vectorizes to b128 loads.
__device__ __forceinline__ v16bf load_a_rm(const float* A, int lda,
                                           int m_base, int k0) {
  const int t    = threadIdx.x & 31;
  const int half = t >> 4;
  const int lane = t & 15;
  const float* row = A + (size_t)(m_base + lane) * (size_t)lda;
  v16bf f;
#pragma unroll
  for (int e = 0; e < 16; ++e) {
    const int k = k0 + e + 16 * half;
    f[e] = (__bf16)row[k];
  }
  return f;
}

__device__ __forceinline__ float silu_f(float v) {
  return v * (1.0f / (1.0f + __expf(-v)));
}
__device__ __forceinline__ float softplus_f(float v) {
  return v > 20.0f ? v : log1pf(__expf(v));
}

// ------------------- K1: backbone conv + ReLU + mean -----------------------
// grid = BS*2 blocks of 256 threads (8 waves). Wave w handles 16 output
// channels (one 16x16x32 WMMA M-tile); loops over 24 N-tiles of positions.
// B fragment: lanes 0-15 = ch0 taps 0-15 (16 contiguous floats), lanes
// 16-31 = ch1 -> 4x global_load_b128 per lane per tile.
__global__ void bb_conv_kernel(const float* __restrict__ x,
                               const float* __restrict__ W_bb,
                               const float* __restrict__ b_bb,
                               float* __restrict__ feat) {
  const int wave   = threadIdx.x >> 5;
  const int bs     = blockIdx.x >> 1;
  const int m_tile = ((blockIdx.x & 1) << 3) + wave;   // 0..15
  const int m_base = m_tile << 4;
  const int t      = threadIdx.x & 31;
  const int half   = t >> 4;
  const int lane   = t & 15;

  // A fragment: W_bb laid out [oc][ch][tap] = oc*32 + kk, kk = ch*16+tap
  const v16bf a = load_a_rm(W_bb, C_IN * K_BB, m_base, 0);

  float bias_r[8];
#pragma unroll
  for (int r = 0; r < 8; ++r) bias_r[r] = b_bb[m_base + r + 8 * half];

  const float* xb = x + (size_t)bs * (C_IN * T_IN) + (size_t)half * T_IN;

  float sum[8];
#pragma unroll
  for (int r = 0; r < 8; ++r) sum[r] = 0.0f;

  const int n_tiles = (T_OUT + 15) >> 4;   // 24
  for (int nt = 0; nt < n_tiles; ++nt) {
    const int pos = (nt << 4) + lane;      // output position for this lane
    v16bf bfrag;
    if (pos < T_OUT) {
      const float* p = xb + pos * STRIDE_BB;       // 16 contiguous taps
      __builtin_prefetch(p + 256, 0, 1);           // global_prefetch_b8
#pragma unroll
      for (int e = 0; e < 16; ++e) bfrag[e] = (__bf16)p[e];
    } else {
#pragma unroll
      for (int e = 0; e < 16; ++e) bfrag[e] = (__bf16)0.0f;
    }
    v8f c = {};
    c = __builtin_amdgcn_wmma_f32_16x16x32_bf16(false, a, false, bfrag,
                                                (short)0, c, false, false);
    const bool valid = pos < T_OUT;
#pragma unroll
    for (int r = 0; r < 8; ++r) {
      float v = c[r] + bias_r[r];
      v = v > 0.0f ? v : 0.0f;             // ReLU
      if (valid) sum[r] += v;
    }
  }

  // reduce across the 16 N-lanes of each half (mean over positions)
#pragma unroll
  for (int r = 0; r < 8; ++r) {
#pragma unroll
    for (int m = 8; m >= 1; m >>= 1) sum[r] += __shfl_xor(sum[r], m, 16);
  }
  if (lane == 0) {
    const float inv = 1.0f / (float)T_OUT;
#pragma unroll
    for (int r = 0; r < 8; ++r)
      feat[(size_t)bs * D_MODEL + m_base + r + 8 * half] = sum[r] * inv;
  }
}

// ------------- LDS-staged WMMA bf16 GEMM: C = A(MxK) * B(KxN) --------------
// Block = 128x128 C macro-tile, 8 waves. Wave w owns rows [w*16, w*16+16)
// and 8 v8f accumulators (one per 16-col N-tile). Per K-step of 32 the block
// cooperatively stages the 32x128 fp32 B slab into LDS as bf16 column-major,
// so each lane's B fragment (col n, K-half h) is one contiguous 32B LDS read.
// Requirements: M % 128 == 0, K % 32 == 0. N arbitrary (zero-padded).
__global__ void gemm_bf16_kernel(const float* __restrict__ A,
                                 const float* __restrict__ B,
                                 float* __restrict__ C,
                                 int M, int N, int K) {
  __shared__ __align__(128) __bf16 Bs[128 * 32];   // [col][k] bf16, 8 KB

  const int NB  = (N + 127) >> 7;
  const int mb  = blockIdx.x / NB;
  const int nb  = blockIdx.x - mb * NB;
  const int n0  = nb << 7;
  const int rem = N - n0;                          // columns in this slab
  const int nt_count = rem >= 128 ? 8 : ((rem + 15) >> 4);   // block-uniform

  const int wave   = threadIdx.x >> 5;
  const int m_base = (mb << 7) + (wave << 4);
  const int t      = threadIdx.x & 31;
  const int half   = t >> 4;
  const int lane   = t & 15;

  // staging assignment: thread -> (k row, 16-col group)
  const int skk = threadIdx.x >> 3;                // 0..31
  const int sc0 = (threadIdx.x & 7) << 4;          // 0,16,...,112

  v8f acc[8];
#pragma unroll
  for (int i = 0; i < 8; ++i) acc[i] = (v8f){};

  for (int k0 = 0; k0 < K; k0 += 32) {
    __syncthreads();                               // prior reads complete
    {
      const float* src = B + (size_t)(k0 + skk) * (size_t)N + n0 + sc0;
      if (rem >= 128) {                            // fast path: vectorizes
#pragma unroll
        for (int j = 0; j < 16; ++j)
          Bs[(sc0 + j) * 32 + skk] = (__bf16)src[j];
      } else {
#pragma unroll
        for (int j = 0; j < 16; ++j) {
          const float v = (sc0 + j < rem) ? src[j] : 0.0f;
          Bs[(sc0 + j) * 32 + skk] = (__bf16)v;
        }
      }
    }
    __syncthreads();                               // slab visible

    const v16bf a = load_a_rm(A, K, m_base, k0);
#pragma unroll
    for (int nt = 0; nt < 8; ++nt) {
      if (nt < nt_count) {                         // block-uniform guard
        const v16bf b =
            *(const v16bf*)&Bs[((nt << 4) + lane) * 32 + (half << 4)];
        acc[nt] = __builtin_amdgcn_wmma_f32_16x16x32_bf16(
            false, a, false, b, (short)0, acc[nt], false, false);
      }
    }
  }

  // store: acc[nt][r] -> row m_base + r + 8*half, col n0 + nt*16 + lane
#pragma unroll
  for (int nt = 0; nt < 8; ++nt) {
    if (nt < nt_count) {
      const int n = n0 + (nt << 4) + lane;
      if (n < N) {
#pragma unroll
        for (int r = 0; r < 8; ++r)
          C[(size_t)(m_base + r + 8 * half) * (size_t)N + n] = acc[nt][r];
      }
    }
  }
}

// ------------------- K3: causal depthwise conv(4) + bias + silu ------------
__global__ void dconv_kernel(const float* __restrict__ xz,     // (BS, 2*D_INNER)
                             const float* __restrict__ Wc,     // (D_INNER,1,4)
                             const float* __restrict__ bc,
                             float* __restrict__ xm_act) {     // (BS, D_INNER)
  const int idx = blockIdx.x * blockDim.x + threadIdx.x;
  if (idx >= BS * D_INNER) return;
  const int row = idx >> 9;          // (b,s)
  const int d   = idx & (D_INNER - 1);
  const int b   = row >> 9;
  const int s   = row & (SEQ - 1);
  float acc = bc[d];
#pragma unroll
  for (int j = 0; j < 4; ++j) {
    const int ss = s - 3 + j;
    if (ss >= 0)
      acc += xz[(size_t)((b << 9) + ss) * (2 * D_INNER) + d] * Wc[d * 4 + j];
  }
  xm_act[idx] = silu_f(acc);
}

// ------------------- K5: dt = softplus(proj[:, :16] @ W_dt + b_dt) ---------
__global__ void dt_kernel(const float* __restrict__ proj,      // (BS, 48)
                          const float* __restrict__ W_dt,      // (16, 512)
                          const float* __restrict__ b_dt,
                          float* __restrict__ dt) {            // (BS, 512)
  const int idx = blockIdx.x * blockDim.x + threadIdx.x;
  if (idx >= BS * D_INNER) return;
  const int row = idx >> 9;
  const int d   = idx & (D_INNER - 1);
  const float* pr = proj + (size_t)row * NPROJ;
  float acc = b_dt[d];
#pragma unroll
  for (int r = 0; r < DT_RANK; ++r) acc += pr[r] * W_dt[r * D_INNER + d];
  dt[idx] = softplus_f(acc);
}

// ------------------- K6: selective scan (16 lanes per (b,d) pair) ----------
__global__ void scan_kernel(const float* __restrict__ xm,      // (BS, 512)
                            const float* __restrict__ dt,      // (BS, 512)
                            const float* __restrict__ proj,    // (BS, 48)
                            const float* __restrict__ A_log,   // (512, 16)
                            float* __restrict__ y) {           // (BS, 512)
  const int pair = blockIdx.x * 16 + (threadIdx.x >> 4);  // (b,d), 2048 total
  const int n    = threadIdx.x & 15;                      // state index
  const int b    = pair >> 9;
  const int d    = pair & (D_INNER - 1);
  const float A_n = -__expf(A_log[d * D_STATE + n]);
  float h = 0.0f;
  for (int s = 0; s < SEQ; ++s) {
    const int row = (b << 9) + s;
    const float dt_v = dt[(size_t)row * D_INNER + d];
    const float x_v  = xm[(size_t)row * D_INNER + d];
    const float B_v  = proj[(size_t)row * NPROJ + DT_RANK + n];
    const float C_v  = proj[(size_t)row * NPROJ + DT_RANK + D_STATE + n];
    h = __expf(dt_v * A_n) * h + dt_v * B_v * x_v;
    float contrib = h * C_v;
#pragma unroll
    for (int m = 8; m >= 1; m >>= 1) contrib += __shfl_xor(contrib, m, 16);
    if (n == 0) y[(size_t)row * D_INNER + d] = contrib;
  }
}

// ------------------- K7: y = (y + xm*D) * silu(z) --------------------------
__global__ void gate_kernel(const float* __restrict__ xm_act,
                            const float* __restrict__ xz,
                            const float* __restrict__ Dp,
                            float* __restrict__ y) {
  const int idx = blockIdx.x * blockDim.x + threadIdx.x;
  if (idx >= BS * D_INNER) return;
  const int row = idx >> 9;
  const int d   = idx & (D_INNER - 1);
  const float z = xz[(size_t)row * (2 * D_INNER) + D_INNER + d];
  y[idx] = (y[idx] + xm_act[idx] * Dp[d]) * silu_f(z);
}

// ------------------- K9: classifier (N=5) ----------------------------------
__global__ void cls_kernel(const float* __restrict__ hidden,   // (BS, 256)
                           const float* __restrict__ W_cls,    // (256, 5)
                           const float* __restrict__ b_cls,
                           float* __restrict__ out) {          // (BS, 5)
  const int idx = blockIdx.x * blockDim.x + threadIdx.x;
  if (idx >= BS * N_CLS) return;
  const int row = idx / N_CLS;
  const int c   = idx - row * N_CLS;
  const float* h = hidden + (size_t)row * D_MODEL;
  float acc = b_cls[c];
#pragma unroll 4
  for (int k = 0; k < D_MODEL; ++k) acc += h[k] * W_cls[k * N_CLS + c];
  out[idx] = acc;
}

// ---------------------------------------------------------------------------
extern "C" void kernel_launch(void* const* d_in, const int* in_sizes, int n_in,
                              void* d_out, int out_size, void* d_ws, size_t ws_size,
                              hipStream_t stream) {
  (void)in_sizes; (void)n_in; (void)out_size; (void)ws_size;
  const float* x       = (const float*)d_in[0];
  const float* W_bb    = (const float*)d_in[1];
  const float* b_bb    = (const float*)d_in[2];
  const float* W_in    = (const float*)d_in[3];
  const float* W_convd = (const float*)d_in[4];
  const float* b_convd = (const float*)d_in[5];
  const float* W_xp    = (const float*)d_in[6];
  const float* W_dt    = (const float*)d_in[7];
  const float* b_dt    = (const float*)d_in[8];
  const float* A_log   = (const float*)d_in[9];
  const float* Dp      = (const float*)d_in[10];
  const float* W_out   = (const float*)d_in[11];
  const float* W_cls   = (const float*)d_in[12];
  const float* b_cls   = (const float*)d_in[13];
  float* out = (float*)d_out;

  // workspace layout (floats)
  float* ws      = (float*)d_ws;
  float* feat    = ws;                              // 2048*256
  float* xz      = feat   + (size_t)BS * D_MODEL;   // 2048*1024
  float* xm_act  = xz     + (size_t)BS * 2 * D_INNER;
  float* proj    = xm_act + (size_t)BS * D_INNER;   // 2048*48
  float* dt      = proj   + (size_t)BS * NPROJ;     // 2048*512
  float* y       = dt     + (size_t)BS * D_INNER;   // 2048*512
  float* hidden  = y      + (size_t)BS * D_INNER;   // 2048*256

  const int TPB = 256;

  // K1: backbone conv + relu + mean -> feat
  bb_conv_kernel<<<BS * 2, TPB, 0, stream>>>(x, W_bb, b_bb, feat);

  // K2: xz = feat @ W_in   (M=2048, N=1024, K=256) -> 16*8 = 128 blocks
  gemm_bf16_kernel<<<(BS / 128) * ((2 * D_INNER + 127) / 128), TPB, 0, stream>>>(
      feat, W_in, xz, BS, 2 * D_INNER, D_MODEL);

  // K3: depthwise causal conv + silu -> xm_act
  dconv_kernel<<<(BS * D_INNER + TPB - 1) / TPB, TPB, 0, stream>>>(
      xz, W_convd, b_convd, xm_act);

  // K4: proj = xm_act @ W_xp   (M=2048, N=48, K=512) -> 16*1 = 16 blocks
  gemm_bf16_kernel<<<(BS / 128) * ((NPROJ + 127) / 128), TPB, 0, stream>>>(
      xm_act, W_xp, proj, BS, NPROJ, D_INNER);

  // K5: dt
  dt_kernel<<<(BS * D_INNER + TPB - 1) / TPB, TPB, 0, stream>>>(
      proj, W_dt, b_dt, dt);

  // K6: selective scan -> y
  scan_kernel<<<BS / 16, TPB, 0, stream>>>(xm_act, dt, proj, A_log, y);

  // K7: gate
  gate_kernel<<<(BS * D_INNER + TPB - 1) / TPB, TPB, 0, stream>>>(
      xm_act, xz, Dp, y);

  // K8: hidden = y @ W_out   (M=2048, N=256, K=512) -> 16*2 = 32 blocks
  gemm_bf16_kernel<<<(BS / 128) * ((D_MODEL + 127) / 128), TPB, 0, stream>>>(
      y, W_out, hidden, BS, D_MODEL, D_INNER);

  // K9: classifier
  cls_kernel<<<(BS * N_CLS + TPB - 1) / TPB, TPB, 0, stream>>>(
      hidden, W_cls, b_cls, out);
}
